// AttentionBlock_10118942949685
// MI455X (gfx1250) — compile-verified, matrix operand order checked by reading, not run on hardware
//
#include <hip/hip_runtime.h>
#include <hip/hip_bf16.h>

typedef __attribute__((ext_vector_type(16))) _Float16 v16h;
typedef __attribute__((ext_vector_type(8)))  float    v8f;
typedef unsigned int u32x4 __attribute__((ext_vector_type(4)));
typedef int          i32x4 __attribute__((ext_vector_type(4)));
typedef int          i32x8 __attribute__((ext_vector_type(8)));

#define BATCH 8
#define C_CH  128
#define HW    4096
#define NGRP  8

#if __has_builtin(__builtin_amdgcn_tensor_load_to_lds)
#define HAS_TDM 1
#else
#define HAS_TDM 0
#endif

// ---------------------------------------------------------------------------
// WMMA helpers (CDNA5 gfx1250, wave32)
// ---------------------------------------------------------------------------
__device__ __forceinline__ v8f wmma_f32_f16(v16h a, v16h b, v8f c) {
  return __builtin_amdgcn_wmma_f32_16x16x32_f16(false, a, false, b, (short)0, c,
                                                false, false);
}

// Load a 16x32 f16 A- or B-fragment from row-major memory (ISA 7.12.2):
// lane m = lane&15, half = lane>>4; frag[j] = K(half*8+j), frag[8+j] = K(16+half*8+j)
__device__ __forceinline__ v16h load_frag16x32(const _Float16* __restrict__ base,
                                               int ld, int lane) {
  const int m    = lane & 15;
  const int half = (lane >> 4) & 1;
  const _Float16* r = base + (size_t)m * ld + half * 8;
  v16h f;
#pragma unroll
  for (int j = 0; j < 8; ++j) f[j] = r[j];
#pragma unroll
  for (int j = 0; j < 8; ++j) f[8 + j] = r[16 + j];
  return f;
}

// 32-bit LDS byte offset of a generic pointer that points into LDS.
__device__ __forceinline__ unsigned lds_off(const void* p) {
  return (unsigned)(unsigned long long)
         (__attribute__((address_space(3))) const char*)p;
}

// ---------------------------------------------------------------------------
// Tensor Data Mover: 2-D tile load (data_size = 8 bytes) per ISA ch.8 D#.
// elems_per_row / row_stride are in 8-byte units.  rows==1 -> flat copy.
// 6-arg builtin form (clang-23 toolchain): (g0, g1, g2, g3, g4, cpol).
// ---------------------------------------------------------------------------
__device__ __forceinline__ void tdm_load_2d(unsigned lds_byte_off,
                                            const void* gaddr,
                                            unsigned elems_per_row,
                                            unsigned rows,
                                            unsigned row_stride_elems) {
#if HAS_TDM
  unsigned long long ga = (unsigned long long)(size_t)gaddr;
  u32x4 g0;
  g0[0] = 1u;                                            // count=1 (valid user D#)
  g0[1] = lds_byte_off;                                  // lds_addr
  g0[2] = (unsigned)ga;                                  // global_addr[31:0]
  g0[3] = (unsigned)((ga >> 32) & 0x1FFFFFFull)          // global_addr[56:32]
        | (2u << 30);                                    // type = 2 ("image")
  i32x8 g1;
  g1[0] = (3 << 16);                                     // data_size=3 (8B), no mask
  g1[1] = (int)((elems_per_row & 0xFFFFu) << 16);        // tensor_dim0[15:0]
  g1[2] = (int)((elems_per_row >> 16) |
                ((rows & 0xFFFFu) << 16));               // dim0[31:16] | tensor_dim1[15:0]
  g1[3] = (int)((rows >> 16) | (elems_per_row << 16));   // dim1[31:16] | tile_dim0
  g1[4] = (int)rows;                                     // tile_dim1 (tile_dim2=0)
  g1[5] = (int)row_stride_elems;                         // tensor_dim0_stride[31:0]
  g1[6] = 0;                                             // stride0[47:32] | stride1[15:0]
  g1[7] = 0;                                             // stride1[47:16]
  i32x4 gz  = {0, 0, 0, 0};
  i32x8 gz8 = {0, 0, 0, 0, 0, 0, 0, 0};
  __builtin_amdgcn_tensor_load_to_lds(g0, g1, gz, gz, gz8, 0);
#endif
}

// ---------------------------------------------------------------------------
// 1) GroupNorm statistics: one block per (batch, group)
// ---------------------------------------------------------------------------
__global__ void gn_stats_kernel(const float* __restrict__ x,
                                float* __restrict__ mu, float* __restrict__ rs) {
  const int bg = blockIdx.x;
  const int b = bg >> 3, g = bg & 7;
  const float* base = x + ((size_t)b * C_CH + g * 16) * HW;
  float s = 0.f, ss = 0.f;
  for (int i = threadIdx.x; i < 16 * HW; i += blockDim.x) {
    float v = base[i];
    s += v; ss += v * v;
  }
  __shared__ float sh_s[256], sh_q[256];
  sh_s[threadIdx.x] = s; sh_q[threadIdx.x] = ss;
  __syncthreads();
  for (int o = 128; o > 0; o >>= 1) {
    if ((int)threadIdx.x < o) {
      sh_s[threadIdx.x] += sh_s[threadIdx.x + o];
      sh_q[threadIdx.x] += sh_q[threadIdx.x + o];
    }
    __syncthreads();
  }
  if (threadIdx.x == 0) {
    float m   = sh_s[0] * (1.f / 65536.f);
    float var = sh_q[0] * (1.f / 65536.f) - m * m;
    mu[bg] = m;
    rs[bg] = rsqrtf(var + 1e-5f);
  }
}

// ---------------------------------------------------------------------------
// 2) Apply GroupNorm, transpose (B,C,HW) -> (B,HW,C), downconvert to f16
// ---------------------------------------------------------------------------
__global__ void gn_apply_kernel(const float* __restrict__ x,
                                const float* __restrict__ gamma,
                                const float* __restrict__ beta,
                                const float* __restrict__ mu,
                                const float* __restrict__ rs,
                                _Float16* __restrict__ xh) {
  const int bg = blockIdx.x;
  const int b = bg >> 3, g = bg & 7;
  const float m = mu[bg], r = rs[bg];
  const float* base = x + ((size_t)b * C_CH + g * 16) * HW;
  for (int i = threadIdx.x; i < 16 * HW; i += blockDim.x) {
    int cc = i >> 12;
    int t  = i & (HW - 1);
    int c  = g * 16 + cc;
    float v = (base[i] - m) * r * gamma[c] + beta[c];
    xh[((size_t)b * HW + t) * C_CH + c] = (_Float16)v;
  }
}

// ---------------------------------------------------------------------------
// 3) Convert weight matrices f32 -> f16 (wq|wk|wv|wo contiguous)
// ---------------------------------------------------------------------------
__global__ void cvt_w_kernel(const float* __restrict__ wq,
                             const float* __restrict__ wk,
                             const float* __restrict__ wv,
                             const float* __restrict__ wo,
                             _Float16* __restrict__ w16) {
  int i = blockIdx.x * blockDim.x + threadIdx.x;
  if (i < C_CH * C_CH) {
    w16[i]             = (_Float16)wq[i];
    w16[16384 + i]     = (_Float16)wk[i];
    w16[2 * 16384 + i] = (_Float16)wv[i];
    w16[3 * 16384 + i] = (_Float16)wo[i];
  }
}

// ---------------------------------------------------------------------------
// 4) QKV projection GEMM: one wave per 16-token strip, full 128 channels.
//    Q,K stored (B,HW,C) f16; V stored transposed (B,C,HW) f16.
// ---------------------------------------------------------------------------
__global__ void qkv_kernel(const _Float16* __restrict__ xh,
                           const _Float16* __restrict__ w16,
                           const float* __restrict__ bq,
                           const float* __restrict__ bk,
                           const float* __restrict__ bv,
                           _Float16* __restrict__ Q,
                           _Float16* __restrict__ K,
                           _Float16* __restrict__ Vt) {
  const int lane = threadIdx.x & 31;
  const int wave = (blockIdx.x * blockDim.x + threadIdx.x) >> 5;
  const int b  = wave >> 8;
  const int t0 = (wave & 255) << 4;
  const int half = (lane >> 4) & 1;
  const int ncol = lane & 15;

  const _Float16* A = xh + ((size_t)b * HW + t0) * C_CH;
  v16h af[4];
#pragma unroll
  for (int c = 0; c < 4; ++c) af[c] = load_frag16x32(A + c * 32, C_CH, lane);

  const float* biases[3] = {bq, bk, bv};
#pragma unroll
  for (int wsel = 0; wsel < 3; ++wsel) {
    const _Float16* W = w16 + wsel * 16384;
#pragma unroll
    for (int n = 0; n < 8; ++n) {
      v8f acc = {};
#pragma unroll
      for (int c = 0; c < 4; ++c) {
        v16h bf = load_frag16x32(W + (size_t)(n * 16) * C_CH + c * 32, C_CH, lane);
        acc = wmma_f32_f16(af[c], bf, acc);
      }
      const int ccol = n * 16 + ncol;
      const float bias = biases[wsel][ccol];
#pragma unroll
      for (int rr = 0; rr < 8; ++rr) {
        const int t = t0 + half * 8 + rr;
        const float v = acc[rr] + bias;
        if (wsel == 0)      Q[((size_t)b * HW + t) * C_CH + ccol]  = (_Float16)v;
        else if (wsel == 1) K[((size_t)b * HW + t) * C_CH + ccol]  = (_Float16)v;
        else                Vt[((size_t)b * C_CH + ccol) * HW + t] = (_Float16)v;
      }
    }
  }
}

// ---------------------------------------------------------------------------
// 5) Flash attention, workgroup-cooperative:
//    8 waves / block own 128 queries of one batch; 32-key K/V chunks are
//    staged into LDS with the Tensor Data Mover (double-buffered, DMA
//    overlapped with WMMA), TENSORcnt-tracked.  Per chunk per wave:
//    8 WMMAs for S=QK^T, shfl_xor online softmax, P via per-wave LDS tile,
//    8 WMMAs for O += P V.
//
//    Dynamic LDS layout (40960 B):
//      [0..16384)      K tiles  2 x (32 keys x 128 ch)  f16
//      [16384..32768)  V tiles  2 x (128 ch x 32 keys)  f16
//      [32768..40960)  P tiles  8 waves x (16 x 32)     f16
//    NOTE: all smem-derived pointers/offsets are computed as scalar runtime
//    expressions (no brace-initialized pointer arrays -> no addrspacecast
//    static initializers for ld.lld to choke on).
// ---------------------------------------------------------------------------
__global__ void __launch_bounds__(256)
flash_kernel(const _Float16* __restrict__ Q,
             const _Float16* __restrict__ K,
             const _Float16* __restrict__ Vt,
             _Float16* __restrict__ Aout) {
  extern __shared__ char smem[];
  const int lane = threadIdx.x & 31;
  const int wv   = threadIdx.x >> 5;
  _Float16* ptile = (_Float16*)(smem + 32768) + wv * (16 * 32);
  const unsigned lds_base = lds_off(smem);

  const int blk  = blockIdx.x;            // 256 blocks: 32 per batch
  const int b    = blk >> 5;
  const int q0   = (blk & 31) * 128;      // first query of this block
  const int t0   = q0 + wv * 16;          // this wave's 16-query strip
  const int half = (lane >> 4) & 1;
  const int ncol = lane & 15;
  const float scale = 0.08838834764831845f;   // 1/sqrt(128)

  const _Float16* Qb = Q  + ((size_t)b * HW + t0) * C_CH;
  const _Float16* Kb = K  + (size_t)b * HW * C_CH;
  const _Float16* Vb = Vt + (size_t)b * C_CH * HW;

  v16h qf[4];
#pragma unroll
  for (int c = 0; c < 4; ++c) qf[c] = load_frag16x32(Qb + c * 32, C_CH, lane);

  const v8f vzero = {};
  v8f O[8];
  float M[8], L[8];
#pragma unroll
  for (int n = 0; n < 8; ++n) O[n] = vzero;
#pragma unroll
  for (int r = 0; r < 8; ++r) { M[r] = -1e30f; L[r] = 0.f; }

  // ---- staging helpers (BUF is a runtime 0/1 expression) ---------------
#define STAGE_CHUNK(J, BUF)                                                    \
  do {                                                                         \
    const unsigned kofs_ = lds_base + (unsigned)(BUF) * 8192u;                 \
    const unsigned vofs_ = lds_base + 16384u + (unsigned)(BUF) * 8192u;        \
    if (HAS_TDM) {                                                             \
      if (wv == 0) {                                                           \
        /* K chunk: contiguous 8 KB (32 keys x 128 ch) */                      \
        tdm_load_2d(kofs_, Kb + (size_t)(J) * C_CH, 1024u, 1u, 1024u);         \
        /* V chunk: 128 rows x 64 B, row stride 8192 B  */                     \
        tdm_load_2d(vofs_, Vb + (J), 8u, 128u, 1024u);                         \
      }                                                                        \
    } else {                                                                   \
      const int tid = threadIdx.x;                                             \
      unsigned long long* kd =                                                 \
          (unsigned long long*)(smem + (size_t)(BUF) * 8192);                  \
      const unsigned long long* ks =                                           \
          (const unsigned long long*)(Kb + (size_t)(J) * C_CH);                \
      for (int i = tid; i < 1024; i += 256) kd[i] = ks[i];                     \
      unsigned long long* vd =                                                 \
          (unsigned long long*)(smem + 16384 + (size_t)(BUF) * 8192);          \
      for (int i = tid; i < 1024; i += 256) {                                  \
        int rr = i >> 3, ww = i & 7;                                           \
        vd[i] = ((const unsigned long long*)(Vb + (size_t)rr * HW + (J)))[ww]; \
      }                                                                        \
    }                                                                          \
  } while (0)

#define WAIT_CHUNK()                                                           \
  do {                                                                         \
    if (HAS_TDM) {                                                             \
      if (wv == 0) __builtin_amdgcn_s_wait_tensorcnt(0);                       \
    }                                                                          \
    __syncthreads();                                                           \
  } while (0)

  // ---- pipeline: prefetch chunk 0, then overlap DMA(j+32) with compute(j)
  int cur = 0;
  STAGE_CHUNK(0, cur);
  WAIT_CHUNK();

  for (int j = 0; j < HW; j += 32) {
    if (j + 32 < HW) STAGE_CHUNK(j + 32, cur ^ 1);

    const _Float16* Kt = (const _Float16*)(smem + (size_t)cur * 8192);
    const _Float16* Vl = (const _Float16*)(smem + 16384 + (size_t)cur * 8192);

    // --- S tile: 16 queries x 32 keys from LDS ---
    v8f s0 = vzero, s1 = vzero;
#pragma unroll
    for (int c = 0; c < 4; ++c) {
      v16h k0 = load_frag16x32(Kt + c * 32,              C_CH, lane);
      v16h k1 = load_frag16x32(Kt + 16 * C_CH + c * 32,  C_CH, lane);
      s0 = wmma_f32_f16(qf[c], k0, s0);
      s1 = wmma_f32_f16(qf[c], k1, s1);
    }

    // --- online softmax (C/D layout: lane owns rows half*8+r, col ncol) ---
    float alpha[8];
#pragma unroll
    for (int r = 0; r < 8; ++r) {
      float v0 = s0[r] * scale, v1 = s1[r] * scale;
      float mx = fmaxf(v0, v1);
#pragma unroll
      for (int off = 1; off < 16; off <<= 1)
        mx = fmaxf(mx, __shfl_xor(mx, off, 32));
      float mnew = fmaxf(M[r], mx);
      float p0 = __expf(v0 - mnew), p1 = __expf(v1 - mnew);
      float ls = p0 + p1;
#pragma unroll
      for (int off = 1; off < 16; off <<= 1)
        ls += __shfl_xor(ls, off, 32);
      alpha[r] = __expf(M[r] - mnew);
      L[r] = L[r] * alpha[r] + ls;
      M[r] = mnew;
      const int row = half * 8 + r;
      ptile[row * 32 + ncol]      = (_Float16)p0;
      ptile[row * 32 + 16 + ncol] = (_Float16)p1;
    }
#pragma unroll
    for (int n = 0; n < 8; ++n)
#pragma unroll
      for (int r = 0; r < 8; ++r) O[n][r] *= alpha[r];

    // --- O += P @ V (both operands from LDS) ---
    v16h pf = load_frag16x32(ptile, 32, lane);
#pragma unroll
    for (int n = 0; n < 8; ++n) {
      v16h vf = load_frag16x32(Vl + (n * 16) * 32, 32, lane);
      O[n] = wmma_f32_f16(pf, vf, O[n]);
    }

    __syncthreads();                 // everyone done reading buffer `cur`
    if (j + 32 < HW) WAIT_CHUNK();   // next buffer landed
    cur ^= 1;
  }

  // --- finalize: divide by row sums, store f16 (B,HW,C) ---
#pragma unroll
  for (int n = 0; n < 8; ++n) {
    const int ccol = n * 16 + ncol;
#pragma unroll
    for (int r = 0; r < 8; ++r) {
      const int t = t0 + half * 8 + r;
      Aout[((size_t)b * HW + t) * C_CH + ccol] = (_Float16)(O[n][r] / L[r]);
    }
  }
#undef STAGE_CHUNK
#undef WAIT_CHUNK
}

// ---------------------------------------------------------------------------
// 6) Output projection + bias + residual -> (B,C,H,W) f32
// ---------------------------------------------------------------------------
__global__ void proj_kernel(const _Float16* __restrict__ Aout,
                            const _Float16* __restrict__ w16,
                            const float* __restrict__ bo,
                            const float* __restrict__ x,
                            float* __restrict__ out) {
  const int lane = threadIdx.x & 31;
  const int wave = (blockIdx.x * blockDim.x + threadIdx.x) >> 5;
  const int b  = wave >> 8;
  const int t0 = (wave & 255) << 4;
  const int half = (lane >> 4) & 1;
  const int ncol = lane & 15;

  const _Float16* A = Aout + ((size_t)b * HW + t0) * C_CH;
  const _Float16* W = w16 + 3 * 16384;
  v16h af[4];
#pragma unroll
  for (int c = 0; c < 4; ++c) af[c] = load_frag16x32(A + c * 32, C_CH, lane);

#pragma unroll
  for (int n = 0; n < 8; ++n) {
    v8f acc = {};
#pragma unroll
    for (int c = 0; c < 4; ++c) {
      v16h bf = load_frag16x32(W + (size_t)(n * 16) * C_CH + c * 32, C_CH, lane);
      acc = wmma_f32_f16(af[c], bf, acc);
    }
    const int ccol = n * 16 + ncol;
    const float bias = bo[ccol];
    const size_t basei = ((size_t)b * C_CH + ccol) * HW + t0 + half * 8;
#pragma unroll
    for (int r = 0; r < 8; ++r)
      out[basei + r] = acc[r] + bias + x[basei + r];
  }
}

// ---------------------------------------------------------------------------
// Host launch
// ---------------------------------------------------------------------------
extern "C" void kernel_launch(void* const* d_in, const int* in_sizes, int n_in,
                              void* d_out, int out_size, void* d_ws, size_t ws_size,
                              hipStream_t stream) {
  const float* x     = (const float*)d_in[0];
  const float* gamma = (const float*)d_in[1];
  const float* beta  = (const float*)d_in[2];
  const float* wq    = (const float*)d_in[3];
  const float* bq    = (const float*)d_in[4];
  const float* wk    = (const float*)d_in[5];
  const float* bk    = (const float*)d_in[6];
  const float* wv    = (const float*)d_in[7];
  const float* bv    = (const float*)d_in[8];
  const float* wo    = (const float*)d_in[9];
  const float* bo    = (const float*)d_in[10];
  float* out = (float*)d_out;

  char* ws = (char*)d_ws;
  const size_t TOK = (size_t)BATCH * HW;
  const size_t F16MAT = TOK * C_CH * sizeof(_Float16);   // 8 MB
  _Float16* xh  = (_Float16*)(ws);
  _Float16* Qb  = (_Float16*)(ws + F16MAT);
  _Float16* Kb  = (_Float16*)(ws + 2 * F16MAT);
  _Float16* Vtb = (_Float16*)(ws + 3 * F16MAT);
  _Float16* Ao  = (_Float16*)(ws + 4 * F16MAT);
  _Float16* w16 = (_Float16*)(ws + 5 * F16MAT);
  float*    mu  = (float*)(ws + 5 * F16MAT + 4 * 16384 * sizeof(_Float16));
  float*    rs  = mu + BATCH * NGRP;

  gn_stats_kernel<<<BATCH * NGRP, 256, 0, stream>>>(x, mu, rs);
  gn_apply_kernel<<<BATCH * NGRP, 256, 0, stream>>>(x, gamma, beta, mu, rs, xh);
  cvt_w_kernel<<<(C_CH * C_CH + 255) / 256, 256, 0, stream>>>(wq, wk, wv, wo, w16);

  qkv_kernel <<<256, 256, 0, stream>>>(xh, w16, bq, bk, bv, Qb, Kb, Vtb);
  flash_kernel<<<256, 256, 40960, stream>>>(Qb, Kb, Vtb, Ao);
  proj_kernel<<<256, 256, 0, stream>>>(Ao, w16, bo, x, out);
}
